// EquivariantAttention_66056597012501
// MI455X (gfx1250) — matrix operand order repaired and checked by reading, not compiled, not run on hardware
//
#include <hip/hip_runtime.h>
#include <hip/hip_bf16.h>
#include <math.h>

// Problem constants (match reference setup_inputs)
#define BB    2
#define NPTS  2048
#define DD    256
#define EE    16
#define HH    8
#define DHD   64
#define KNB   64
#define INNER 512    // HH*DHD
#define QKVW  1536   // 3*INNER

typedef __attribute__((ext_vector_type(16))) __bf16 v16bf;
typedef __attribute__((ext_vector_type(8)))  float  v8f;

__device__ __forceinline__ unsigned int pack_bf16x2(float lo, float hi) {
  unsigned int ulo = __float_as_uint(lo);
  ulo = (ulo + 0x7FFFu + ((ulo >> 16) & 1u)) >> 16;
  unsigned int uhi = __float_as_uint(hi);
  uhi = (uhi + 0x7FFFu + ((uhi >> 16) & 1u)) >> 16;
  return ulo | (uhi << 16);
}

__device__ __forceinline__ float gelu_exact(float x) {
  return 0.5f * x * (1.0f + erff(x * 0.70710678118654752f));
}

// ---------------------------------------------------------------------------
// Kernel 1: pairwise distances + top-64 neighbor selection (one block per row)
// ---------------------------------------------------------------------------
__global__ void __launch_bounds__(256)
topk_kernel(const float* __restrict__ coors, int* __restrict__ nbidx,
            float* __restrict__ rdist, float* __restrict__ rcoors) {
  __shared__ float sdist[NPTS];
  __shared__ float rv[256];
  __shared__ int   ri[256];
  const int tid = threadIdx.x;
  const int bi  = blockIdx.x;
  const int b   = bi / NPTS;
  const float cx = coors[(size_t)bi * 3 + 0];
  const float cy = coors[(size_t)bi * 3 + 1];
  const float cz = coors[(size_t)bi * 3 + 2];
  const float* cb = coors + (size_t)b * NPTS * 3;
  for (int j = tid; j < NPTS; j += 256) {
    float dx = cx - cb[j * 3 + 0];
    float dy = cy - cb[j * 3 + 1];
    float dz = cz - cb[j * 3 + 2];
    sdist[j] = sqrtf(dx * dx + dy * dy + dz * dz);
  }
  __syncthreads();
  for (int sel = 0; sel < KNB; ++sel) {
    float bv = 3.4e38f; int bj = NPTS;
    for (int j = tid; j < NPTS; j += 256) {
      float v = sdist[j];
      if (v < bv) { bv = v; bj = j; }   // strided scan keeps lowest index on tie
    }
    rv[tid] = bv; ri[tid] = bj;
    __syncthreads();
    for (int s = 128; s > 0; s >>= 1) {
      if (tid < s) {
        float v2 = rv[tid + s]; int i2 = ri[tid + s];
        if (v2 < rv[tid] || (v2 == rv[tid] && i2 < ri[tid])) { rv[tid] = v2; ri[tid] = i2; }
      }
      __syncthreads();
    }
    if (tid == 0) {
      int jm = ri[0];
      nbidx[(size_t)bi * KNB + sel] = jm;
      rdist[(size_t)bi * KNB + sel] = rv[0];
      rcoors[((size_t)bi * KNB + sel) * 3 + 0] = cx - cb[jm * 3 + 0];
      rcoors[((size_t)bi * KNB + sel) * 3 + 1] = cy - cb[jm * 3 + 1];
      rcoors[((size_t)bi * KNB + sel) * 3 + 2] = cz - cb[jm * 3 + 2];
      sdist[jm] = 3.4e38f;
    }
    __syncthreads();
  }
}

// ---------------------------------------------------------------------------
// Kernel 2: C[M,N] = A[M,K] * B[K,N] (+bias) with bf16 WMMA, f32 accumulate.
// Block = 128 threads = 4 waves. Block tile = 16M x 256N; each wave owns a
// 16M x 64N strip = 4 accumulators. Per K-step the wave loads its A fragment
// once and issues 4 independent back-to-back WMMAs (fills the XDL pipe and
// amortizes the LDS/barrier cost 4x vs one-WMMA-per-barrier).
// ---------------------------------------------------------------------------
__global__ void __launch_bounds__(128)
wmma_gemm_bf16(const float* __restrict__ A, const float* __restrict__ B,
               const float* __restrict__ bias, float* __restrict__ C,
               int M, int N, int K) {
  __shared__ unsigned int sA[16 * 16];        // 16 rows x 32 bf16 (packed pairs)
  __shared__ unsigned int sB[4][64 * 16];     // per-wave: 64 cols x 32 bf16
  const int tid  = threadIdx.x;
  const int wave = tid >> 5;
  const int lane = tid & 31;
  const int tm = blockIdx.y << 4;
  const int tn = (blockIdx.x << 8) + (wave << 6);   // wave's 64-col strip base
  v8f acc[4] = {v8f{}, v8f{}, v8f{}, v8f{}};
  for (int k0 = 0; k0 < K; k0 += 32) {
    // Stage A tile (shared by all 4 waves): 256 packed uints.
    for (int e = tid; e < 256; e += 128) {
      int r = e >> 4, cp = e & 15;
      const float* s = A + (size_t)(tm + r) * K + k0 + (cp << 1);
      sA[e] = pack_bf16x2(s[0], s[1]);
    }
    // Prefetch next A tile rows into the cache hierarchy.
    if (k0 + 32 < K) __builtin_prefetch(A + (size_t)(tm + (tid & 15)) * K + k0 + 32, 0, 1);
    // Stage per-wave B strip (32K x 64N) column-major: 1024 packed uints.
    for (int e = lane; e < 1024; e += 32) {
      int col = e >> 4, kp = e & 15;
      const float* s0 = B + (size_t)(k0 + (kp << 1)) * N + tn + col;
      sB[wave][e] = pack_bf16x2(s0[0], s0[N]);
    }
    __syncthreads();
    union { v16bf v; unsigned int d[8]; } fa, fb0, fb1, fb2, fb3;
    const int mr = lane & 15, hh = lane >> 4;
#pragma unroll
    for (int p = 0; p < 8; ++p) {
      // A layout: VGPR p holds K = ((p&4)?16:0) + hh*8 + 2*(p&3) (+1)
      int kbA = ((p & 4) << 2) + (hh << 3) + ((p & 3) << 1);
      fa.d[p] = sA[(mr << 4) + (kbA >> 1)];
      // B layout: VGPR p holds K = ((p&4)?16:0) + hh*8 + 2*(p&3) (+1)
      int bidx = ((p & 4) << 1) + (hh << 2) + (p & 3);
      fb0.d[p] = sB[wave][((0 << 4) + mr) * 16 + bidx];
      fb1.d[p] = sB[wave][((1 << 4) + mr) * 16 + bidx];
      fb2.d[p] = sB[wave][((2 << 4) + mr) * 16 + bidx];
      fb3.d[p] = sB[wave][((3 << 4) + mr) * 16 + bidx];
    }
    // Four independent WMMAs sharing the A fragment.
    acc[0] = __builtin_amdgcn_wmma_f32_16x16x32_bf16(false, fa.v, false, fb0.v,
                                                     (short)0, acc[0], false, false);
    acc[1] = __builtin_amdgcn_wmma_f32_16x16x32_bf16(false, fa.v, false, fb1.v,
                                                     (short)0, acc[1], false, false);
    acc[2] = __builtin_amdgcn_wmma_f32_16x16x32_bf16(false, fa.v, false, fb2.v,
                                                     (short)0, acc[2], false, false);
    acc[3] = __builtin_amdgcn_wmma_f32_16x16x32_bf16(false, fa.v, false, fb3.v,
                                                     (short)0, acc[3], false, false);
    __syncthreads();
  }
  const int col = lane & 15, hh = lane >> 4;
#pragma unroll
  for (int nb = 0; nb < 4; ++nb) {
    const int cbase = tn + (nb << 4) + col;
    const float bvv = bias ? bias[cbase] : 0.0f;
#pragma unroll
    for (int r = 0; r < 8; ++r) {
      int m = tm + r + (hh << 3);   // C layout: VGPR r -> M=r (lanes 0-15), M=r+8 (16-31)
      C[(size_t)m * N + cbase] = acc[nb][r] + bvv;
    }
  }
}

// ---------------------------------------------------------------------------
// Kernel 3: attention core. One block (256 thr) per (b,i) query.
// Gathers neighbor K/V, applies rotary, edge MLP, dual softmax, coordinate
// branch (-> coors_delta) and feature aggregation (-> out_feat).
// ---------------------------------------------------------------------------
__global__ void __launch_bounds__(256)
attn_core(const float* __restrict__ qkv,
          const int*   __restrict__ nbidx,
          const float* __restrict__ rdist,
          const float* __restrict__ rcoors,
          const float* __restrict__ edges,
          const float* __restrict__ W_e1, const float* __restrict__ b_e1,
          const float* __restrict__ W_e2, const float* __restrict__ b_e2,
          const float* __restrict__ W_c,  const float* __restrict__ b_c,
          const float* __restrict__ W_g,  const float* __restrict__ b_g,
          const float* __restrict__ comb, const float* __restrict__ cscale,
          float* __restrict__ out_feat, float* __restrict__ out_coors) {
  __shared__ float qs[INNER];
  __shared__ int   nbs[KNB];
  __shared__ float rds[KNB];
  __shared__ float rcs[KNB * 3];
  __shared__ float cosb[KNB * 32];
  __shared__ float sinb[KNB * 32];
  __shared__ float We1[24 * 48];
  __shared__ float Be1[48];
  __shared__ float We2[48 * 8];
  __shared__ float Be2[8];
  __shared__ float Wc[64], Bc[8], Wg[64], Bg[8], Cmb[8];
  __shared__ float qkraw[KNB * HH];
  __shared__ float mjh[KNB * HH];
  __shared__ float coorw[KNB * HH];
  __shared__ float gatew[KNB * HH];
  __shared__ float attnw[KNB * HH];
  __shared__ float relouts[24];
  __shared__ float vbuf[8 * INNER];
  __shared__ float sscale;

  const int tid = threadIdx.x;
  const int bi  = blockIdx.x;
  const int b   = bi / NPTS;
  const int i   = bi % NPTS;

  for (int e = tid; e < 24 * 48; e += 256) We1[e] = W_e1[e];
  for (int e = tid; e < 48 * 8;  e += 256) We2[e] = W_e2[e];
  if (tid < 48) Be1[tid] = b_e1[tid];
  if (tid < 8)  { Be2[tid] = b_e2[tid]; Bc[tid] = b_c[tid]; Bg[tid] = b_g[tid]; Cmb[tid] = comb[tid]; }
  if (tid < 64) { Wc[tid] = W_c[tid]; Wg[tid] = W_g[tid]; }
  if (tid == 0) sscale = cscale[0];

  const float* qrow = qkv + (size_t)bi * QKVW;
  for (int e = tid; e < INNER; e += 256) qs[e] = qrow[e];
  if (tid < KNB) {
    nbs[tid] = nbidx[(size_t)bi * KNB + tid];
    rds[tid] = rdist[(size_t)bi * KNB + tid];
  }
  for (int e = tid; e < KNB * 3; e += 256) rcs[e] = rcoors[(size_t)bi * KNB * 3 + e];
  __syncthreads();

  // Rotary tables: freqs repeated in pairs; 32 unique per neighbor.
  for (int e = tid; e < KNB * 32; e += 256) {
    int j = e >> 5, t = e & 31;
    float tt = fminf(rds[j] * 100.0f, 5000.0f);
    float fr = tt * __expf(-(float)(2 * t) * (1.0f / 64.0f) * 9.210340371976184f); // 10000^-(2t/64)
    cosb[e] = __cosf(fr);
    sinb[e] = __sinf(fr);
  }
  __syncthreads();

  // qk dots: stage rotated K rows for 8 neighbors at a time through LDS.
  for (int jc = 0; jc < KNB; jc += 8) {
    for (int pe = tid; pe < 8 * (INNER / 2); pe += 256) {
      int jj = pe >> 8, pr = pe & 255;      // pr = pair index within 512
      int j = jc + jj;
      const float* krow = qkv + ((size_t)b * NPTS + nbs[j]) * QKVW + INNER;
      int t = pr & 31;
      float c = cosb[(j << 5) + t], sn = sinb[(j << 5) + t];
      float x0 = krow[2 * pr], x1 = krow[2 * pr + 1];
      vbuf[jj * INNER + 2 * pr]     = x0 * c - x1 * sn;
      vbuf[jj * INNER + 2 * pr + 1] = x1 * c + x0 * sn;
    }
    __syncthreads();
    if (tid < 64) {
      int jj = tid >> 3, h = tid & 7;
      const float* kr = vbuf + jj * INNER + h * DHD;
      const float* qr = qs + h * DHD;
      float s = 0.0f;
#pragma unroll 8
      for (int d = 0; d < DHD; ++d) s += qr[d] * kr[d];
      qkraw[(jc + jj) * HH + h] = s;      // no 1/sqrt(dh): edges present
    }
    __syncthreads();
  }

  // Edge MLP per neighbor: [qk(8) ; edges(16)] -> 48 (gelu) -> 8
  if (tid < KNB) {
    int j = tid;
    float x[24];
#pragma unroll
    for (int h = 0; h < 8; ++h) x[h] = qkraw[j * HH + h];
    const float* er = edges + (((size_t)b * NPTS + i) * NPTS + nbs[j]) * EE;
#pragma unroll
    for (int c = 0; c < EE; ++c) x[8 + c] = er[c];
    float hid[48];
    for (int o = 0; o < 48; ++o) {
      float s = Be1[o];
      for (int c = 0; c < 24; ++c) s += x[c] * We1[c * 48 + o];
      hid[o] = gelu_exact(s);
    }
    for (int o = 0; o < 8; ++o) {
      float s = Be2[o];
      for (int c = 0; c < 48; ++c) s += hid[c] * We2[c * 8 + o];
      mjh[j * HH + o] = s;
    }
  }
  __syncthreads();

  // Coordinate weights (gelu -> W_c) and gate (tanh(W_g))
  for (int e = tid; e < KNB * HH; e += 256) {
    int j = e >> 3, h = e & 7;
    float cw = Bc[h], gw = Bg[h];
#pragma unroll
    for (int c = 0; c < 8; ++c) {
      float mv = mjh[j * HH + c];
      cw += gelu_exact(mv) * Wc[c * 8 + h];
      gw += mv * Wg[c * 8 + h];
    }
    coorw[e] = cw;
    gatew[e] = tanhf(gw);
  }
  __syncthreads();

  // Per-head softmaxes over neighbors (j).
  if (tid < 8) {
    int h = tid;
    float mx = -3.4e38f;
    for (int j = 0; j < KNB; ++j) mx = fmaxf(mx, coorw[j * HH + h]);
    float sum = 0.0f;
    for (int j = 0; j < KNB; ++j) { float e2 = __expf(coorw[j * HH + h] - mx); coorw[j * HH + h] = e2; sum += e2; }
    float inv = 1.0f / sum;
    for (int j = 0; j < KNB; ++j) coorw[j * HH + h] *= inv;
  } else if (tid < 16) {
    int h = tid - 8;
    float mx = -3.4e38f;
    for (int j = 0; j < KNB; ++j) mx = fmaxf(mx, mjh[j * HH + h]);
    float sum = 0.0f;
    for (int j = 0; j < KNB; ++j) { float e2 = __expf(mjh[j * HH + h] - mx); attnw[j * HH + h] = e2; sum += e2; }
    float inv = 1.0f / sum;
    for (int j = 0; j < KNB; ++j) attnw[j * HH + h] *= inv;
  }
  __syncthreads();

  // Coordinate branch -> coors_delta (3 floats per query)
  if (tid < 24) {
    int c = tid >> 3, h = tid & 7;
    float s = 0.0f;
    for (int j = 0; j < KNB; ++j) {
      float nx = rcs[j * 3], ny = rcs[j * 3 + 1], nz = rcs[j * 3 + 2];
      float nrm = fmaxf(sqrtf(nx * nx + ny * ny + nz * nz), 1e-8f);
      float rn = rcs[j * 3 + c] / nrm * sscale;
      s += coorw[j * HH + h] * gatew[j * HH + h] * rn;
    }
    relouts[c * 8 + h] = s;
  }
  __syncthreads();
  if (tid < 3) {
    float s = 0.0f;
#pragma unroll
    for (int h = 0; h < 8; ++h) s += relouts[tid * 8 + h] * Cmb[h];
    out_coors[(size_t)bi * 3 + tid] = s;
  }

  // Feature branch: out[h,d] = sum_j attn[j,h] * rot(v)[j,h,d]
  float acc0 = 0.0f, acc1 = 0.0f;
  const int h0 = tid >> 6,        d0 = tid & 63;
  const int h1 = (tid + 256) >> 6, d1 = tid & 63;
  for (int jc = 0; jc < KNB; jc += 8) {
    __syncthreads();
    for (int pe = tid; pe < 8 * (INNER / 2); pe += 256) {
      int jj = pe >> 8, pr = pe & 255;
      int j = jc + jj;
      const float* vrow = qkv + ((size_t)b * NPTS + nbs[j]) * QKVW + 2 * INNER;
      int t = pr & 31;
      float c = cosb[(j << 5) + t], sn = sinb[(j << 5) + t];
      float x0 = vrow[2 * pr], x1 = vrow[2 * pr + 1];
      vbuf[jj * INNER + 2 * pr]     = x0 * c - x1 * sn;
      vbuf[jj * INNER + 2 * pr + 1] = x1 * c + x0 * sn;
    }
    __syncthreads();
#pragma unroll
    for (int jj = 0; jj < 8; ++jj) {
      acc0 += attnw[(jc + jj) * HH + h0] * vbuf[jj * INNER + h0 * DHD + d0];
      acc1 += attnw[(jc + jj) * HH + h1] * vbuf[jj * INNER + h1 * DHD + d1];
    }
  }
  out_feat[(size_t)bi * INNER + h0 * DHD + d0] = acc0;
  out_feat[(size_t)bi * INNER + h1 * DHD + d1] = acc1;
}

// ---------------------------------------------------------------------------
extern "C" void kernel_launch(void* const* d_in, const int* in_sizes, int n_in,
                              void* d_out, int out_size, void* d_ws, size_t ws_size,
                              hipStream_t stream) {
  const float* feats  = (const float*)d_in[0];
  const float* coors  = (const float*)d_in[1];
  const float* edges  = (const float*)d_in[2];
  const float* W_qkv  = (const float*)d_in[3];
  const float* W_out  = (const float*)d_in[4];
  const float* b_out  = (const float*)d_in[5];
  const float* W_e1   = (const float*)d_in[6];
  const float* b_e1   = (const float*)d_in[7];
  const float* W_e2   = (const float*)d_in[8];
  const float* b_e2   = (const float*)d_in[9];
  const float* W_c    = (const float*)d_in[10];
  const float* b_c    = (const float*)d_in[11];
  const float* W_g    = (const float*)d_in[12];
  const float* b_g    = (const float*)d_in[13];
  const float* comb   = (const float*)d_in[14];
  const float* cscale = (const float*)d_in[15];

  const int rows = BB * NPTS;   // 4096
  char* ws = (char*)d_ws;
  float* qkv      = (float*)ws; ws += (size_t)rows * QKVW * sizeof(float);
  float* out_feat = (float*)ws; ws += (size_t)rows * INNER * sizeof(float);
  int*   nbidx    = (int*)ws;   ws += (size_t)rows * KNB * sizeof(int);
  float* rdist    = (float*)ws; ws += (size_t)rows * KNB * sizeof(float);
  float* rcoors   = (float*)ws; ws += (size_t)rows * KNB * 3 * sizeof(float);

  float* node_out  = (float*)d_out;                       // (b,n,256)
  float* coors_out = node_out + (size_t)rows * DD;        // (b,n,3)

  // 1) neighbors
  topk_kernel<<<rows, 256, 0, stream>>>(coors, nbidx, rdist, rcoors);

  // 2) QKV projection: (4096x256)x(256x1536), WMMA bf16, 256-col block tiles
  dim3 g1(QKVW / 256, rows / 16);
  wmma_gemm_bf16<<<g1, 128, 0, stream>>>(feats, W_qkv, nullptr, qkv,
                                         rows, QKVW, DD);

  // 3) attention core (writes coors_delta directly, features to workspace)
  attn_core<<<rows, 256, 0, stream>>>(qkv, nbidx, rdist, rcoors, edges,
                                      W_e1, b_e1, W_e2, b_e2, W_c, b_c,
                                      W_g, b_g, comb, cscale,
                                      out_feat, coors_out);

  // 4) output projection: (4096x512)x(512x256)+b, WMMA bf16
  dim3 g2(DD / 256, rows / 16);
  wmma_gemm_bf16<<<g2, 128, 0, stream>>>(out_feat, W_out, b_out, node_out,
                                         rows, DD, INNER);
}
